// MultiHeadSelfAttention_81432579932238
// MI455X (gfx1250) — matrix operand order repaired and checked by reading, not compiled, not run on hardware
//
#include <hip/hip_runtime.h>
#include <hip/hip_bf16.h>

#define N_EMBD 1024
#define N_HEADS 16
#define HEAD 64
#define BATCH 2
#define SEQ 2048
#define MTOT (BATCH * SEQ)   /* 4096 rows */
#define QKV_N (3 * N_EMBD)   /* 3072 */

typedef __attribute__((ext_vector_type(16))) _Float16 v16h;
typedef __attribute__((ext_vector_type(8)))  _Float16 v8h;
typedef __attribute__((ext_vector_type(4)))  _Float16 v4h;
typedef __attribute__((ext_vector_type(8)))  float    v8f;

// Padded LDS layout for a 64(N) x 1024(K) f16 B-panel:
// 8 f16 of padding after every 512 f16 -> row stride 1040 f16 (2080 B).
// element (n,k) at  n*1040 + k + (k>>9)*8   (f16 units)
#define LDSB_ROWSTRIDE 1040
#define LDSB_BYTES (64 * LDSB_ROWSTRIDE * 2)

#if defined(__has_builtin)
#if __has_builtin(__builtin_amdgcn_tensor_load_to_lds) && \
    __has_builtin(__builtin_amdgcn_s_wait_tensorcnt)
#define HAVE_TDM 1
#endif
#if __has_builtin(__builtin_amdgcn_sched_barrier)
#define SCHED_BARRIER() __builtin_amdgcn_sched_barrier(0)
#endif
#endif
#ifndef HAVE_TDM
#define HAVE_TDM 0
#endif
#ifndef SCHED_BARRIER
#define SCHED_BARRIER() asm volatile("" ::: "memory")
#endif

#if HAVE_TDM
typedef __attribute__((ext_vector_type(4))) unsigned int u32x4;
typedef __attribute__((ext_vector_type(8))) int i32x8;
typedef __attribute__((ext_vector_type(4))) int i32x4;
#endif

// ---------------------------------------------------------------------------
// WMMA fragment loaders (contiguous k = 16*half + s, same order for A and B)
// A (16x32, MxK), row-major source [M][K]:
//   lanes 0-15: M=lane, K = 0..15 (contiguous)
//   lanes 16-31: M=lane-16, K = 16..31 (contiguous)
// ---------------------------------------------------------------------------
__device__ __forceinline__ v16h load_a_frag(const _Float16* base, int lda) {
  const int lane = threadIdx.x & 31;
  const int l = lane & 15, hf = lane >> 4;
  const _Float16* p = base + (size_t)l * lda + hf * 16;
  v8h lo = *(const v8h*)(p);
  v8h hi = *(const v8h*)(p + 8);
  v16h a;
#pragma unroll
  for (int i = 0; i < 8; ++i) { a[i] = lo[i]; a[8 + i] = hi[i]; }
  return a;
}

// B (32x16, KxN) loaded from B^T stored row-major [N][K]:
//   lanes 0-15:  column N=lane,    K = 0..15  (contiguous)
//   lanes 16-31: column N=lane-16, K = 16..31 (contiguous)
__device__ __forceinline__ v16h load_b_frag(const _Float16* baseT, int ldb) {
  const int lane = threadIdx.x & 31;
  const int n = lane & 15, hf = lane >> 4;
  const _Float16* p = baseT + (size_t)n * ldb + hf * 16;
  v8h lo = *(const v8h*)(p);
  v8h hi = *(const v8h*)(p + 8);
  v16h b;
#pragma unroll
  for (int i = 0; i < 8; ++i) { b[i] = lo[i]; b[8 + i] = hi[i]; }
  return b;
}

#define WMMA_F16(a, b, c) \
  __builtin_amdgcn_wmma_f32_16x16x32_f16(false, (a), false, (b), (short)0, (c), false, false)

// ---------------------------------------------------------------------------
// Stage one 64x1024 f16 B-panel (rows n0..n0+63 of srcT[N][1024]) into padded
// LDS.  TDM path: one Tensor-DMA descriptor (2D tile, pad_interval=256 dwords,
// pad_amount=4 dwords reproduces the padded layout), issued by wave 0,
// completion via s_wait_tensorcnt then a workgroup barrier.
// ---------------------------------------------------------------------------
__device__ __forceinline__ void stage_b_panel(const _Float16* __restrict__ srcT,
                                              int n0, _Float16* lds) {
#if HAVE_TDM
  if ((threadIdx.x >> 5) == 0) {
    const unsigned long long ga =
        (unsigned long long)(uintptr_t)(srcT + (size_t)n0 * N_EMBD);
    const unsigned int la = (unsigned int)(uintptr_t)lds;  // low 32 bits = LDS offset
    u32x4 g0;
    g0[0] = 1u;                                  // count=1 valid user descriptor
    g0[1] = la;                                  // lds_addr
    g0[2] = (unsigned int)ga;                    // global_addr[31:0]
    g0[3] = (unsigned int)((ga >> 32) & 0x01FFFFFFull) | 0x80000000u;  // [56:32] | type=2
    i32x8 g1;
    g1[0] = (1 << 16)    // data_size = 2 bytes
          | (1 << 20)    // pad_enable
          | (7 << 22)    // pad_interval: 256 dwords
          | (3 << 25);   // pad_amount: 4 dwords (8 f16)
    g1[1] = (int)((1024u & 0xffffu) << 16);          // tensor_dim0[15:0]
    g1[2] = (int)((1024u >> 16) | (3072u << 16));    // tensor_dim0[31:16] | tensor_dim1[15:0]
    g1[3] = (int)((3072u >> 16) | (1024u << 16));    // tensor_dim1[31:16] | tile_dim0=1024
    g1[4] = 64;                                      // tile_dim1=64, tile_dim2=0
    g1[5] = 1024;                                    // tensor_dim0_stride[31:0]
    g1[6] = 0;
    g1[7] = 0;
    i32x4 gz = {0, 0, 0, 0};
    i32x8 gz8 = {0, 0, 0, 0, 0, 0, 0, 0};
    __builtin_amdgcn_tensor_load_to_lds(g0, g1, gz, gz, gz8, 0);
    __builtin_amdgcn_s_wait_tensorcnt(0);
  }
  __syncthreads();
#else
  const int tid = threadIdx.x;
  for (int idx = tid; idx < 64 * 64; idx += 256) {
    const int row = idx >> 6;
    const int k = (idx & 63) * 16;
    *(v8h*)(lds + (size_t)row * LDSB_ROWSTRIDE + k + ((k >> 9) << 3)) =
        *(const v8h*)(srcT + (size_t)(n0 + row) * N_EMBD + k);
  }
  __syncthreads();
#endif
}

// ---------------------------------------------------------------------------
// Precision conversion / weight transpose kernels
// ---------------------------------------------------------------------------
__global__ void cvt_f32_to_f16_kernel(const float* __restrict__ src,
                                      _Float16* __restrict__ dst, int n) {
  int i = (blockIdx.x * blockDim.x + threadIdx.x) * 4;
  if (i + 3 < n) {
    float4 v = *(const float4*)(src + i);
    v4h o;
    o[0] = (_Float16)v.x; o[1] = (_Float16)v.y;
    o[2] = (_Float16)v.z; o[3] = (_Float16)v.w;
    *(v4h*)(dst + i) = o;
  }
}

// LDS-tiled transpose-convert: wt[n][k] = (f16) w[k][n]; w is [rows=K][cols=N].
// Grid: (cols/32, rows/32), block 256 (32x8). Coalesced both directions.
__global__ __launch_bounds__(256) void transpose_w_kernel(
    const float* __restrict__ w, _Float16* __restrict__ wt, int rows, int cols) {
  __shared__ float tile[32][33];
  const int bx = blockIdx.x * 32;  // n base
  const int by = blockIdx.y * 32;  // k base
  const int tx = threadIdx.x & 31, ty = threadIdx.x >> 5;
#pragma unroll
  for (int i = 0; i < 4; ++i)
    tile[ty + 8 * i][tx] = w[(size_t)(by + ty + 8 * i) * cols + bx + tx];
  __syncthreads();
#pragma unroll
  for (int i = 0; i < 4; ++i)
    wt[(size_t)(bx + ty + 8 * i) * rows + by + tx] = (_Float16)tile[tx][ty + 8 * i];
}

// ---------------------------------------------------------------------------
// Kernel 1: QKV projection  xh[4096x1024] @ Wqkv[1024x3072] + bqkv
// 8 waves/block; block tile M=256 x N=64; whole B-panel resident in LDS
// (barrier-free k-loop).  A sched_barrier between the fragment loads and the
// WMMA phase forces one ds_load clause into 4 distinct octets -> single
// s_wait_dscnt, 8 back-to-back WMMAs, no WAR-hazard NOPs.
// ---------------------------------------------------------------------------
__global__ __launch_bounds__(256) void qkv_gemm_kernel(
    const _Float16* __restrict__ xh, const _Float16* __restrict__ wqkvT,
    const float* __restrict__ bqkv, _Float16* __restrict__ Q,
    _Float16* __restrict__ K, _Float16* __restrict__ Vt) {
  extern __shared__ _Float16 ldsB[];
  const int n0 = blockIdx.x * 64;
  const int m0 = blockIdx.y * 256;
  const int tid = threadIdx.x;
  const int wid = tid >> 5;
  const int lane = tid & 31;
  const int ln = lane & 15, hf = lane >> 4;
  const int mw = m0 + wid * 32;

  stage_b_panel(wqkvT, n0, ldsB);

  const v8f zero = {0.f, 0.f, 0.f, 0.f, 0.f, 0.f, 0.f, 0.f};
  v8f acc[2][4];
#pragma unroll
  for (int i = 0; i < 2; ++i)
#pragma unroll
    for (int j = 0; j < 4; ++j) acc[i][j] = zero;

  const int hf16 = hf * 16;
  const _Float16* brow = ldsB + (size_t)ln * LDSB_ROWSTRIDE;

#pragma unroll 2
  for (int k0 = 0; k0 < N_EMBD; k0 += 32) {
    v16h a0 = load_a_frag(xh + (size_t)mw * N_EMBD + k0, N_EMBD);
    v16h a1 = load_a_frag(xh + (size_t)(mw + 16) * N_EMBD + k0, N_EMBD);
    const int kk = k0 + hf16;
    const int koff = kk + ((kk >> 9) << 3);
    v16h bfr[4];
#pragma unroll
    for (int nt = 0; nt < 4; ++nt) {
      const _Float16* p = brow + nt * (16 * LDSB_ROWSTRIDE) + koff;
      v8h lo = *(const v8h*)p;
      v8h hi = *(const v8h*)(p + 8);
#pragma unroll
      for (int i = 0; i < 8; ++i) { bfr[nt][i] = lo[i]; bfr[nt][8 + i] = hi[i]; }
    }
    SCHED_BARRIER();  // keep all loads above, all WMMAs below
#pragma unroll
    for (int nt = 0; nt < 4; ++nt) {
      acc[0][nt] = WMMA_F16(a0, bfr[nt], acc[0][nt]);
      acc[1][nt] = WMMA_F16(a1, bfr[nt], acc[1][nt]);
    }
  }

#pragma unroll
  for (int nt = 0; nt < 4; ++nt) {
    const int n = n0 + nt * 16 + ln;
    const float bias = bqkv[n];
    const int h = n / 192;    // head
    const int rem = n % 192;  // 0..63=q, 64..127=k, 128..191=v
#pragma unroll
    for (int mt = 0; mt < 2; ++mt) {
#pragma unroll
      for (int r = 0; r < 8; ++r) {
        const int m = mw + mt * 16 + hf * 8 + r;
        const int bb = m >> 11;          // / SEQ
        const int t = m & (SEQ - 1);
        const _Float16 v = (_Float16)(acc[mt][nt][r] + bias);
        if (rem < 64) {
          Q[((size_t)(bb * N_HEADS + h) * SEQ + t) * HEAD + rem] = v;
        } else if (rem < 128) {
          K[((size_t)(bb * N_HEADS + h) * SEQ + t) * HEAD + (rem - 64)] = v;
        } else {
          Vt[((size_t)(bb * N_HEADS + h) * HEAD + (rem - 128)) * SEQ + t] = v;
        }
      }
    }
  }
}

// ---------------------------------------------------------------------------
// Kernel 2: flash attention, one wave per (b, h, 16-query tile).
// Computes S^T = K*Q^T (per-query softmax is per-lane), online softmax,
// O^T += V^T * P^T.  P^T -> B-fragment via 16 lane-swap shuffles (no LDS).
// K and V^T fragments all load at the top of each key-block (one clause)
// so the global loads overlap the softmax VALU work.
// ---------------------------------------------------------------------------
__global__ __launch_bounds__(256) void attn_kernel(
    const _Float16* __restrict__ Q, const _Float16* __restrict__ K,
    const _Float16* __restrict__ Vt, _Float16* __restrict__ attn) {
  const int tid = threadIdx.x;
  const int lane = tid & 31;
  const int ln = lane & 15, hf = lane >> 4;
  const int wave = blockIdx.x * 8 + (tid >> 5);
  const int qt = wave & 127;
  const int h = (wave >> 7) & 15;
  const int b = wave >> 11;
  const int q0 = qt * 16;

  const _Float16* Qbh = Q + (size_t)(b * N_HEADS + h) * SEQ * HEAD;
  const _Float16* Kbh = K + (size_t)(b * N_HEADS + h) * SEQ * HEAD;
  const _Float16* Vbh = Vt + (size_t)(b * N_HEADS + h) * HEAD * SEQ;

  const v16h bq0 = load_b_frag(Qbh + (size_t)q0 * HEAD, HEAD);       // feats 0..31
  const v16h bq1 = load_b_frag(Qbh + (size_t)q0 * HEAD + 32, HEAD);  // feats 32..63

  const v8f zero = {0.f, 0.f, 0.f, 0.f, 0.f, 0.f, 0.f, 0.f};
  v8f oT[4] = {zero, zero, zero, zero};
  float m_run = -1e30f, l_run = 0.f;
  const int qglob = q0 + ln;
  const int nkb = (q0 + 15) / 32 + 1;  // causal: key blocks of 32

  for (int kb = 0; kb < nkb; ++kb) {
    const int kbase = kb * 32;
    // All K / V^T fragments for this key block up front (one load clause).
    v16h ak0 = load_a_frag(Kbh + (size_t)kbase * HEAD, HEAD);
    v16h ak1 = load_a_frag(Kbh + (size_t)kbase * HEAD + 32, HEAD);
    v16h ak2 = load_a_frag(Kbh + (size_t)(kbase + 16) * HEAD, HEAD);
    v16h ak3 = load_a_frag(Kbh + (size_t)(kbase + 16) * HEAD + 32, HEAD);
    v16h av[4];
#pragma unroll
    for (int ft = 0; ft < 4; ++ft)
      av[ft] = load_a_frag(Vbh + (size_t)(ft * 16) * SEQ + kbase, SEQ);
    SCHED_BARRIER();  // keep the load clause above the compute

    v8f s0 = zero, s1 = zero;  // S^T tiles: [key 0..15 | 16..31][query]
    s0 = WMMA_F16(ak0, bq0, s0);
    s0 = WMMA_F16(ak1, bq1, s0);
    s1 = WMMA_F16(ak2, bq0, s1);
    s1 = WMMA_F16(ak3, bq1, s1);

    // scale 1/sqrt(64), causal mask, per-query (per-lane) block max
    float bm = -1e30f;
#pragma unroll
    for (int r = 0; r < 8; ++r) {
      const int key0 = kbase + hf * 8 + r;
      const int key1 = key0 + 16;
      float v0 = s0[r] * 0.125f;
      float v1 = s1[r] * 0.125f;
      if (key0 > qglob) v0 = -1e30f;
      if (key1 > qglob) v1 = -1e30f;
      s0[r] = v0; s1[r] = v1;
      bm = fmaxf(bm, fmaxf(v0, v1));
    }
    bm = fmaxf(bm, __shfl_xor(bm, 16, 32));  // combine lane n with lane n+16
    const float m_new = fmaxf(m_run, bm);
    const float alpha = __expf(m_run - m_new);
    float ls = 0.f;
#pragma unroll
    for (int r = 0; r < 8; ++r) {
      const float p0 = __expf(s0[r] - m_new);
      const float p1 = __expf(s1[r] - m_new);
      s0[r] = p0; s1[r] = p1;
      ls += p0 + p1;
    }
    ls += __shfl_xor(ls, 16, 32);
    l_run = l_run * alpha + ls;
    m_run = m_new;

    // P^T (D-frag layout) -> B-fragment (32x16) via half-wave swaps
    v16h bp;
#pragma unroll
    for (int r = 0; r < 8; ++r) {
      const float o0 = __shfl_xor(s0[r], 16, 32);
      const float o1 = __shfl_xor(s1[r], 16, 32);
      bp[r]     = (_Float16)(hf ? o1 : s0[r]);  // keys 0..7  / 16..23
      bp[8 + r] = (_Float16)(hf ? s1[r] : o0);  // keys 8..15 / 24..31
    }

    // O^T[feat tile] += V^T(16 feat x 32 key) * P^T(32 key x 16 query)
#pragma unroll
    for (int ft = 0; ft < 4; ++ft) {
#pragma unroll
      for (int r = 0; r < 8; ++r) oT[ft][r] *= alpha;
      oT[ft] = WMMA_F16(av[ft], bp, oT[ft]);
    }
  }

  const float inv_l = 1.0f / l_run;
  _Float16* dst = attn + (size_t)(b * SEQ + q0 + ln) * N_EMBD + h * HEAD;
#pragma unroll
  for (int ft = 0; ft < 4; ++ft) {
    v8h o;
#pragma unroll
    for (int r = 0; r < 8; ++r) o[r] = (_Float16)(oT[ft][r] * inv_l);
    *(v8h*)(dst + ft * 16 + hf * 8) = o;  // 8 contiguous channels per lane
  }
}

// ---------------------------------------------------------------------------
// Kernel 3: output projection  attn[4096x1024] @ Wo[1024x1024] + bo -> f32
// ---------------------------------------------------------------------------
__global__ __launch_bounds__(256) void out_gemm_kernel(
    const _Float16* __restrict__ ah, const _Float16* __restrict__ woT,
    const float* __restrict__ bo, float* __restrict__ out) {
  extern __shared__ _Float16 ldsB[];
  const int n0 = blockIdx.x * 64;
  const int m0 = blockIdx.y * 256;
  const int tid = threadIdx.x;
  const int wid = tid >> 5;
  const int lane = tid & 31;
  const int ln = lane & 15, hf = lane >> 4;
  const int mw = m0 + wid * 32;

  stage_b_panel(woT, n0, ldsB);

  const v8f zero = {0.f, 0.f, 0.f, 0.f, 0.f, 0.f, 0.f, 0.f};
  v8f acc[2][4];
#pragma unroll
  for (int i = 0; i < 2; ++i)
#pragma unroll
    for (int j = 0; j < 4; ++j) acc[i][j] = zero;

  const int hf16 = hf * 16;
  const _Float16* brow = ldsB + (size_t)ln * LDSB_ROWSTRIDE;

#pragma unroll 2
  for (int k0 = 0; k0 < N_EMBD; k0 += 32) {
    v16h a0 = load_a_frag(ah + (size_t)mw * N_EMBD + k0, N_EMBD);
    v16h a1 = load_a_frag(ah + (size_t)(mw + 16) * N_EMBD + k0, N_EMBD);
    const int kk = k0 + hf16;
    const int koff = kk + ((kk >> 9) << 3);
    v16h bfr[4];
#pragma unroll
    for (int nt = 0; nt < 4; ++nt) {
      const _Float16* p = brow + nt * (16 * LDSB_ROWSTRIDE) + koff;
      v8h lo = *(const v8h*)p;
      v8h hi = *(const v8h*)(p + 8);
#pragma unroll
      for (int i = 0; i < 8; ++i) { bfr[nt][i] = lo[i]; bfr[nt][8 + i] = hi[i]; }
    }
    SCHED_BARRIER();  // keep all loads above, all WMMAs below
#pragma unroll
    for (int nt = 0; nt < 4; ++nt) {
      acc[0][nt] = WMMA_F16(a0, bfr[nt], acc[0][nt]);
      acc[1][nt] = WMMA_F16(a1, bfr[nt], acc[1][nt]);
    }
  }

#pragma unroll
  for (int nt = 0; nt < 4; ++nt) {
    const int n = n0 + nt * 16 + ln;
    const float bias = bo[n];
#pragma unroll
    for (int mt = 0; mt < 2; ++mt) {
#pragma unroll
      for (int r = 0; r < 8; ++r) {
        const int m = mw + mt * 16 + hf * 8 + r;
        out[(size_t)m * N_EMBD + n] = acc[mt][nt][r] + bias;
      }
    }
  }
}

// ---------------------------------------------------------------------------
extern "C" void kernel_launch(void* const* d_in, const int* in_sizes, int n_in,
                              void* d_out, int out_size, void* d_ws, size_t ws_size,
                              hipStream_t stream) {
  (void)in_sizes; (void)n_in; (void)out_size; (void)ws_size;
  const float* x    = (const float*)d_in[0];
  const float* Wqkv = (const float*)d_in[1];
  const float* bqkv = (const float*)d_in[2];
  const float* Wo   = (const float*)d_in[3];
  const float* bo   = (const float*)d_in[4];
  float* out = (float*)d_out;

  char* ws = (char*)d_ws;
  size_t off = 0;
  auto carve = [&](size_t bytes) -> char* {
    char* p = ws + off;
    off += (bytes + 255) & ~(size_t)255;
    return p;
  };
  _Float16* xh    = (_Float16*)carve((size_t)MTOT * N_EMBD * 2);
  _Float16* wqkvT = (_Float16*)carve((size_t)QKV_N * N_EMBD * 2);
  _Float16* woT   = (_Float16*)carve((size_t)N_EMBD * N_EMBD * 2);
  _Float16* Qb    = (_Float16*)carve((size_t)MTOT * N_EMBD * 2);
  _Float16* Kb    = (_Float16*)carve((size_t)MTOT * N_EMBD * 2);
  _Float16* Vtb   = (_Float16*)carve((size_t)MTOT * N_EMBD * 2);
  _Float16* ah    = (_Float16*)carve((size_t)MTOT * N_EMBD * 2);

  const int nx = MTOT * N_EMBD;
  cvt_f32_to_f16_kernel<<<(nx / 4 + 255) / 256, 256, 0, stream>>>(x, xh, nx);
  transpose_w_kernel<<<dim3(QKV_N / 32, N_EMBD / 32), 256, 0, stream>>>(
      Wqkv, wqkvT, N_EMBD, QKV_N);
  transpose_w_kernel<<<dim3(N_EMBD / 32, N_EMBD / 32), 256, 0, stream>>>(
      Wo, woT, N_EMBD, N_EMBD);

  qkv_gemm_kernel<<<dim3(QKV_N / 64, MTOT / 256), 256, LDSB_BYTES, stream>>>(
      xh, wqkvT, bqkv, Qb, Kb, Vtb);

  attn_kernel<<<(BATCH * N_HEADS * (SEQ / 16)) / 8, 256, 0, stream>>>(
      Qb, Kb, Vtb, ah);

  out_gemm_kernel<<<dim3(N_EMBD / 64, MTOT / 256), 256, LDSB_BYTES, stream>>>(
      ah, woT, bo, out);
}
